// GATLayer_69355131896600
// MI455X (gfx1250) — matrix operand order repaired and checked by reading, not compiled, not run on hardware
//
#include <hip/hip_runtime.h>

typedef _Float16 v16h __attribute__((ext_vector_type(16)));
typedef float    v8f  __attribute__((ext_vector_type(8)));
typedef float    f32x4 __attribute__((ext_vector_type(4)));

#define NROWS 8192
#define DIM   64
#define NEG_SLOPE 0.01f

// workspace layout (float offsets):
//  [0,64)     v1 = W^T a1
//  [64,128)   v2 = W^T a2
//  [128]      c1 = b.a1    [129] c2 = b.a2    [130] max(s1)
//  [256, 256+8192)          s1
//  [8448, 8448+8192)        s2
//  [16640, 16672)           per-block maxes of s1 (32 blocks)
//  float offset 16704:      xT as _Float16[64][8192]  (x transposed, f16)
#define WS_S1   256
#define WS_S2   8448
#define WS_BMAX 16640
#define WS_XT   16704

__device__ __forceinline__ float lrelu(float t) {
  return t >= 0.0f ? t : NEG_SLOPE * t;
}

// ---- kernel 0: v1,v2,c1,c2 from W, b, a ------------------------------------
__global__ void k_prep(const float* __restrict__ W, const float* __restrict__ b,
                       const float* __restrict__ a, float* __restrict__ ws) {
  int d = threadIdx.x; // 0..63
  float acc1 = 0.0f, acc2 = 0.0f;
  #pragma unroll 8
  for (int k = 0; k < DIM; ++k) {
    float w = W[k * DIM + d];
    acc1 += a[k] * w;
    acc2 += a[DIM + k] * w;
  }
  ws[d] = acc1;
  ws[DIM + d] = acc2;
  if (d == 0) {
    float c1 = 0.0f, c2 = 0.0f;
    for (int k = 0; k < DIM; ++k) {
      c1 += b[k] * a[k];
      c2 += b[k] * a[DIM + k];
    }
    ws[128] = c1;
    ws[129] = c2;
  }
}

// ---- kernel 1: s1, s2, xT (f16 transpose), per-block max(s1) ---------------
__global__ __launch_bounds__(256) void k_rows(const float* __restrict__ x,
                                              float* __restrict__ ws) {
  const float* v1 = ws;
  const float* v2 = ws + DIM;
  float* s1 = ws + WS_S1;
  float* s2 = ws + WS_S2;
  float* bmax = ws + WS_BMAX;
  _Float16* xT = (_Float16*)(ws + WS_XT);

  int tid = threadIdx.x;
  int i = blockIdx.x * 256 + tid;

  float xi[DIM];
  float d1 = ws[128], d2 = ws[129];
  #pragma unroll
  for (int d = 0; d < DIM; ++d) {
    xi[d] = x[(size_t)i * DIM + d];
    d1 += xi[d] * v1[d];
    d2 += xi[d] * v2[d];
  }
  s1[i] = d1;
  s2[i] = d2;
  #pragma unroll
  for (int d = 0; d < DIM; ++d)
    xT[(size_t)d * NROWS + i] = (_Float16)xi[d];

  __shared__ float red[256];
  red[tid] = d1;
  __syncthreads();
  #pragma unroll
  for (int s = 128; s > 0; s >>= 1) {
    if (tid < s) red[tid] = fmaxf(red[tid], red[tid + s]);
    __syncthreads();
  }
  if (tid == 0) bmax[blockIdx.x] = red[0];
}

// ---- kernel 2: reduce 32 block maxes -> max(s1) ----------------------------
__global__ void k_max(float* __restrict__ ws) {
  float m = ws[WS_BMAX + threadIdx.x]; // 32 values, one wave
  #pragma unroll
  for (int off = 16; off > 0; off >>= 1)
    m = fmaxf(m, __shfl_xor(m, off, 32));
  if (threadIdx.x == 0) ws[130] = m;
}

// ---- kernel 3: fused softmax-attention, WMMA P@X ---------------------------
// 128 blocks x 128 threads (4 waves). Wave handles 16 output rows.
// Software-pipelined: next tile's B fragments load while exp/A of the
// current tile computes, so v_exp VALU work hides global-load latency.
__global__ __launch_bounds__(128) void k_main(const float* __restrict__ ws,
                                              float* __restrict__ out) {
  const float* s1g = ws + WS_S1;
  const float* s2g = ws + WS_S2;
  const _Float16* xT = (const _Float16*)(ws + WS_XT);

  __shared__ float s1s[NROWS]; // 32 KB
  int tid = threadIdx.x;
  for (int t = tid; t < NROWS; t += 128) s1s[t] = s1g[t];
  __syncthreads();

  const int wave = tid >> 5;
  const int lane = tid & 31;
  const int m = lane & 15;
  const bool hi = lane >= 16;
  const int rowbase = (blockIdx.x * 4 + wave) * 16;

  const float s2r  = s2g[rowbase + m];       // this lane's row score
  const float s1mx = ws[130];                // global max(s1)
  const float mrow = lrelu(s2r + s1mx);      // exact row max (lrelu monotone)
  const int kb  = hi ? 8 : 0;                // unused: A fragment also uses kb2 as K base
  const int kb2 = hi ? 16 : 0;               // B-fragment K base per half-wave

  v8f C0 = {}, C1 = {}, C2 = {}, C3 = {}, C4 = {};
  v16h Bones;
  #pragma unroll
  for (int e = 0; e < 16; ++e) Bones[e] = (_Float16)1.0f;

  // B fragment base pointers: column n = 16c + m, K contiguous along j
  const _Float16* xT0 = xT + (size_t)( 0 + m) * NROWS + kb2;
  const _Float16* xT1 = xT + (size_t)(16 + m) * NROWS + kb2;
  const _Float16* xT2 = xT + (size_t)(32 + m) * NROWS + kb2;
  const _Float16* xT3 = xT + (size_t)(48 + m) * NROWS + kb2;

  // prologue: load tile 0's B fragments
  v16h B0 = *(const v16h*)(xT0);
  v16h B1 = *(const v16h*)(xT1);
  v16h B2 = *(const v16h*)(xT2);
  v16h B3 = *(const v16h*)(xT3);

  for (int j0 = 0; j0 < NROWS; j0 += 32) {
    // issue next tile's B loads first (wraps to 0 on last iter: always valid,
    // branch-free; result discarded after final iteration)
    const int jn = (j0 + 32) & (NROWS - 1);
    v16h nB0 = *(const v16h*)(xT0 + jn);
    v16h nB1 = *(const v16h*)(xT1 + jn);
    v16h nB2 = *(const v16h*)(xT2 + jn);
    v16h nB3 = *(const v16h*)(xT3 + jn);
    __builtin_prefetch(xT0 + ((j0 + 512) & (NROWS - 1)), 0, 3);

    // Build A fragment (16x32 f16 attention weights) in registers.
    f32x4 q0 = *(const f32x4*)&s1s[j0 + kb2];
    f32x4 q1 = *(const f32x4*)&s1s[j0 + kb2 + 4];
    f32x4 q2 = *(const f32x4*)&s1s[j0 + kb2 + 8];
    f32x4 q3 = *(const f32x4*)&s1s[j0 + kb2 + 12];
    v16h A;
    #pragma unroll
    for (int e = 0; e < 4; ++e) {
      A[e]      = (_Float16)__expf(lrelu(s2r + q0[e]) - mrow);
      A[4 + e]  = (_Float16)__expf(lrelu(s2r + q1[e]) - mrow);
      A[8 + e]  = (_Float16)__expf(lrelu(s2r + q2[e]) - mrow);
      A[12 + e] = (_Float16)__expf(lrelu(s2r + q3[e]) - mrow);
    }

    C0 = __builtin_amdgcn_wmma_f32_16x16x32_f16(false, A, false, B0, (short)0, C0, false, false);
    C1 = __builtin_amdgcn_wmma_f32_16x16x32_f16(false, A, false, B1, (short)0, C1, false, false);
    C2 = __builtin_amdgcn_wmma_f32_16x16x32_f16(false, A, false, B2, (short)0, C2, false, false);
    C3 = __builtin_amdgcn_wmma_f32_16x16x32_f16(false, A, false, B3, (short)0, C3, false, false);
    // rowsum via B = ones: C4[r] holds the rowsum in the same lane/VGPR slot
    C4 = __builtin_amdgcn_wmma_f32_16x16x32_f16(false, A, false, Bones, (short)0, C4, false, false);

    B0 = nB0; B1 = nB1; B2 = nB2; B3 = nB3;
  }

  // C/D layout: VGPR r, lane l -> row = rowbase + r + (hi?8:0), col = 16c + m
  float* ob = out + (size_t)(rowbase + (hi ? 8 : 0)) * DIM + m;
  #pragma unroll
  for (int r = 0; r < 8; ++r) {
    float inv = __builtin_amdgcn_rcpf(C4[r]);  // softmax norm: 1-ulp rcp ok
    ob[(size_t)r * DIM +  0] = C0[r] * inv;
    ob[(size_t)r * DIM + 16] = C1[r] * inv;
    ob[(size_t)r * DIM + 32] = C2[r] * inv;
    ob[(size_t)r * DIM + 48] = C3[r] * inv;
  }
}

extern "C" void kernel_launch(void* const* d_in, const int* in_sizes, int n_in,
                              void* d_out, int out_size, void* d_ws, size_t ws_size,
                              hipStream_t stream) {
  (void)in_sizes; (void)n_in; (void)out_size; (void)ws_size;
  const float* x = (const float*)d_in[0];
  const float* W = (const float*)d_in[1];
  const float* b = (const float*)d_in[2];
  const float* a = (const float*)d_in[3];
  float* ws = (float*)d_ws;
  float* out = (float*)d_out;

  k_prep<<<1, 64, 0, stream>>>(W, b, a, ws);
  k_rows<<<32, 256, 0, stream>>>(x, ws);
  k_max<<<1, 32, 0, stream>>>(ws);
  k_main<<<128, 128, 0, stream>>>(ws, out);
}
